// cLSTMCell_15633680957681
// MI455X (gfx1250) — matrix operand order repaired and checked by reading, not compiled, not run on hardware
//
#include <hip/hip_runtime.h>

typedef unsigned short u16;
typedef __attribute__((ext_vector_type(8)))  u16    v8u;
typedef __attribute__((ext_vector_type(16))) u16    v16u;
typedef __attribute__((ext_vector_type(16))) __bf16 v16bf;
typedef __attribute__((ext_vector_type(8)))  float  v8f;
typedef __attribute__((ext_vector_type(4)))  float  v4f;

union FragU { v16u u; v8u h[2]; };

#define BATCH    8192
#define IN_DIM   1024
#define HID      1024
#define KDIM     2048
#define NGATE    7
#define GATECOLS (NGATE * HID)            /* 7168 */
#define SLAB     ((size_t)BATCH * HID)    /* 8388608 per output */

#define BM   128
#define BN   32
#define BK   32
#define LDSS 40   /* LDS row stride in u16 elems: 32 + 8 pad = 80 B */

__device__ __forceinline__ u16 f32_to_bf16(float f) {
    unsigned int u = __builtin_bit_cast(unsigned int, f);
    u += 0x7FFFu + ((u >> 16) & 1u);   /* round to nearest even */
    return (u16)(u >> 16);
}

__device__ __forceinline__ float sigmoidf_fast(float x) {
    return 1.0f / (1.0f + __expf(-x));
}

/* Async 16B copy global -> LDS, no VGPR staging; tracked by ASYNCcnt. */
__device__ __forceinline__ void async_ld16(unsigned lds_off, const void* gptr) {
    unsigned long long ga = (unsigned long long)(size_t)gptr;
    asm volatile("global_load_async_to_lds_b128 %0, %1, off"
                 :: "v"(lds_off), "v"(ga) : "memory");
}
__device__ __forceinline__ void wait_async0(void) {
    asm volatile("s_wait_asynccnt 0x0" ::: "memory");
}

/* Pack A = [input | hidden] row-major [8192][2048] as bf16. 8 elems/thread. */
__global__ __launch_bounds__(256) void convert_A_kernel(
    const float* __restrict__ inp, const float* __restrict__ hid,
    u16* __restrict__ A) {
    size_t gid  = (size_t)blockIdx.x * blockDim.x + threadIdx.x;
    size_t base = gid * 8;
    size_t b    = base / KDIM;
    int    k    = (int)(base % KDIM);
    const float* src = (k < IN_DIM) ? (inp + b * IN_DIM + k)
                                    : (hid + b * HID + (k - IN_DIM));
    v4f f0 = *(const v4f*)(src);
    v4f f1 = *(const v4f*)(src + 4);
    v8u o;
    o[0] = f32_to_bf16(f0[0]); o[1] = f32_to_bf16(f0[1]);
    o[2] = f32_to_bf16(f0[2]); o[3] = f32_to_bf16(f0[3]);
    o[4] = f32_to_bf16(f1[0]); o[5] = f32_to_bf16(f1[1]);
    o[6] = f32_to_bf16(f1[2]); o[7] = f32_to_bf16(f1[3]);
    *(v8u*)(A + base) = o;
}

/* Transpose-convert W [2048][7168] f32 -> WT [7168][2048] bf16 via LDS tile. */
__global__ __launch_bounds__(256) void convert_WT_kernel(
    const float* __restrict__ W, u16* __restrict__ WT) {
    __shared__ u16 tile[32][33];
    const int j0 = blockIdx.x * 32;
    const int k0 = blockIdx.y * 32;
    const int t  = threadIdx.x;
#pragma unroll
    for (int p = 0; p < 4; ++p) {
        int e = t + 256 * p;
        int kk = e >> 5, jj = e & 31;
        tile[kk][jj] = f32_to_bf16(W[(size_t)(k0 + kk) * GATECOLS + j0 + jj]);
    }
    __syncthreads();
#pragma unroll
    for (int p = 0; p < 4; ++p) {
        int e = t + 256 * p;
        int jj = e >> 5, kk = e & 31;
        WT[(size_t)(j0 + jj) * KDIM + k0 + kk] = tile[kk][jj];
    }
}

/* Fused 7-gate GEMM (bf16 WMMA, f32 accum) + cLSTM epilogue.
   Grid: x = 1024/BN = 32 col tiles, y = 8192/BM = 64 row tiles.
   8 waves as 4(M) x 2(N); each wave: 32 rows x 16 cols x 7 gates = 14 WMMA/K-step.
   Double-buffered LDS filled with global_load_async_to_lds_b128. */
__global__ __launch_bounds__(256) void clstm_fused_kernel(
    const u16*  __restrict__ A,     /* [8192][2048] bf16 */
    const u16*  __restrict__ WT,    /* [7168][2048] bf16 */
    const float* __restrict__ bias, /* [7168] */
    const float* __restrict__ cin,
    const float* __restrict__ ctin,
    float* __restrict__ out) {

    __shared__ __attribute__((aligned(16))) u16 As[2][BM * LDSS];
    __shared__ __attribute__((aligned(16))) u16 Bs[2][NGATE * BN * LDSS];

    const int tid  = threadIdx.x;
    const int wv   = tid >> 5;
    const int lane = tid & 31;
    const int hlf  = lane >> 4;
    const int lrow = lane & 15;
    const int wvM  = wv & 3;     /* row group: 32 rows each   */
    const int wvN  = wv >> 2;    /* col group: 16 cols each   */

    const int tileN = blockIdx.x * BN;
    const int tileM = blockIdx.y * BM;

    /* Per-thread fill coordinates (16B chunks). A: 512 chunks, 2/thread. */
    const int arow0 = tid >> 2, aseg = tid & 3;
    const int arow1 = arow0 + 64;

    v8f acc[2][NGATE] = {};

    /* Issue async fills of one K-tile into LDS buffer `buf`. */
    auto issue_tile = [&](int buf, int k0) {
        async_ld16((unsigned)(size_t)&As[buf][arow0 * LDSS + aseg * 8],
                   A + (size_t)(tileM + arow0) * KDIM + k0 + aseg * 8);
        async_ld16((unsigned)(size_t)&As[buf][arow1 * LDSS + aseg * 8],
                   A + (size_t)(tileM + arow1) * KDIM + k0 + aseg * 8);
        /* B: 7 gates x 32 cols x 64B = 896 chunks = 3.5/thread */
#pragma unroll
        for (int i = 0; i < 4; ++i) {
            if (i == 3 && tid >= 128) break;
            int c = tid + 256 * i;
            int g = c >> 7, rem = c & 127, n = rem >> 2, seg = rem & 3;
            async_ld16((unsigned)(size_t)&Bs[buf][(g * BN + n) * LDSS + seg * 8],
                       WT + (size_t)(g * HID + tileN + n) * KDIM + k0 + seg * 8);
        }
    };

    issue_tile(0, 0);
    int cur = 0;
    for (int k0 = 0; k0 < KDIM; k0 += BK) {
        wait_async0();       /* my async fills for buf `cur` have landed   */
        __syncthreads();     /* ... and everyone else's: tile is visible   */
        if (k0 + BK < KDIM) issue_tile(cur ^ 1, k0 + BK);

        /* A fragments (16x32 bf16): lanes 0-15 K[0,8)+[16,24),
           lanes 16-31 K[8,16)+[24,32) per ISA layout. */
        const u16* ap0 = &As[cur][(wvM * 32 + lrow) * LDSS];
        const u16* ap1 = ap0 + 16 * LDSS;
        FragU ua0, ua1;
        ua0.h[0] = *(const v8u*)(ap0 + hlf * 8);
        ua0.h[1] = *(const v8u*)(ap0 + 16 + hlf * 8);
        ua1.h[0] = *(const v8u*)(ap1 + hlf * 8);
        ua1.h[1] = *(const v8u*)(ap1 + 16 + hlf * 8);
        v16bf a0 = __builtin_bit_cast(v16bf, ua0.u);
        v16bf a1 = __builtin_bit_cast(v16bf, ua1.u);

#pragma unroll
        for (int g = 0; g < NGATE; ++g) {
            /* B fragment (32x16): lane=col, lanes 0-15 K[0,8)+[16,24),
               lanes 16-31 K[8,16)+[24,32), matching A. */
            FragU ub;
            const u16* brow = &Bs[cur][(g * BN + wvN * 16 + lrow) * LDSS];
            ub.h[0] = *(const v8u*)(brow + hlf * 8);
            ub.h[1] = *(const v8u*)(brow + 16 + hlf * 8);
            v16bf bf = __builtin_bit_cast(v16bf, ub.u);
            acc[0][g] = __builtin_amdgcn_wmma_f32_16x16x32_bf16(
                false, a0, false, bf, (short)0, acc[0][g], false, false);
            acc[1][g] = __builtin_amdgcn_wmma_f32_16x16x32_bf16(
                false, a1, false, bf, (short)0, acc[1][g], false, false);
        }
        cur ^= 1;
    }

    /* Epilogue: C/D layout — VGPR r: lanes 0-15 -> M=r, lanes 16-31 -> M=r+8 */
    const int col = tileN + wvN * 16 + lrow;
    float bb[NGATE];
#pragma unroll
    for (int g = 0; g < NGATE; ++g) bb[g] = bias[g * HID + col];

#pragma unroll
    for (int mf = 0; mf < 2; ++mf) {
        const int rowBase = tileM + wvM * 32 + mf * 16 + hlf * 8;
#pragma unroll
        for (int r = 0; r < 8; ++r) {
            const size_t oidx = (size_t)(rowBase + r) * HID + col;
            float p_i  = acc[mf][0][r] + bb[0];
            float p_it = acc[mf][1][r] + bb[1];
            float p_f  = acc[mf][2][r] + bb[2];
            float p_ft = acc[mf][3][r] + bb[3];
            float p_z  = acc[mf][4][r] + bb[4];
            float p_o  = acc[mf][5][r] + bb[5];
            float p_d  = acc[mf][6][r] + bb[6];

            float gi  = sigmoidf_fast(p_i);
            float git = sigmoidf_fast(p_it);
            float gf  = sigmoidf_fast(p_f);
            float gft = sigmoidf_fast(p_ft);
            float z   = 2.0f * sigmoidf_fast(p_z) - 1.0f;
            float go  = sigmoidf_fast(p_o);
            float dl  = fmaxf(p_d, 0.0f) + log1pf(__expf(-fabsf(p_d)));

            out[oidx]            = gf  * cin[oidx]  + gi  * z;
            out[SLAB + oidx]     = gft * ctin[oidx] + git * z;
            out[2 * SLAB + oidx] = go;
            out[3 * SLAB + oidx] = dl;
        }
    }
}

extern "C" void kernel_launch(void* const* d_in, const int* in_sizes, int n_in,
                              void* d_out, int out_size, void* d_ws, size_t ws_size,
                              hipStream_t stream) {
    const float* input  = (const float*)d_in[0];
    const float* hidden = (const float*)d_in[1];
    const float* c      = (const float*)d_in[2];
    const float* ct     = (const float*)d_in[3];
    const float* W      = (const float*)d_in[4];
    const float* b      = (const float*)d_in[5];
    float* out = (float*)d_out;

    /* ws layout: A bf16 [8192*2048] (32 MB) then WT bf16 [7168*2048] (28 MB) */
    u16* A_ws  = (u16*)d_ws;
    u16* WT_ws = A_ws + (size_t)BATCH * KDIM;

    convert_A_kernel<<<(BATCH * KDIM / 8) / 256, 256, 0, stream>>>(input, hidden, A_ws);
    dim3 gridT(GATECOLS / 32, KDIM / 32);
    convert_WT_kernel<<<gridT, 256, 0, stream>>>(W, WT_ws);
    dim3 gridG(HID / BN, BATCH / BM);
    clstm_fused_kernel<<<gridG, 256, 0, stream>>>(A_ws, WT_ws, b, c, ct, out);

    (void)in_sizes; (void)n_in; (void)out_size; (void)ws_size;
}
